// SnailAttentionBlock_14319420965135
// MI455X (gfx1250) — compile-verified, matrix operand order checked by reading, not run on hardware
//
#include <hip/hip_runtime.h>
#include <hip/hip_bf16.h>

typedef __attribute__((ext_vector_type(16))) _Float16 v16h;
typedef __attribute__((ext_vector_type(8)))  _Float16 v8h;
typedef __attribute__((ext_vector_type(8)))  float    v8f;

#define B_ 4
#define C_ 512
#define T_ 4096
#define K_ 256
#define V_ 256

// Workspace layout (bytes). Total ~41 MB.
#define WT_OFF   (0u)          // 3 * 256 * 512 f16  (Wq^T, Wk^T, Wv^T)
#define QH_OFF   (1u << 20)    // B*T*256 f16, rows scaled by 1/16
#define KH_OFF   (9u << 20)    // B*T*256 f16
#define VT_OFF   (17u << 20)   // B*256*T f16 (transposed V)
#define XH_OFF   (25u << 20)   // B*T*512 f16 (x transposed)

// Build a 16x32 f16 A/B fragment (wave32 layout, ISA 7.12.2):
// lane l: idx = l&15, h = l>>4; elems 0..7 = K h*8..h*8+7, elems 8..15 = K 16+h*8..
__device__ __forceinline__ v16h ldfrag(const _Float16* base, int h) {
    v8h lo = *(const v8h*)(base + h * 8);
    v8h hi = *(const v8h*)(base + 16 + h * 8);
    return __builtin_shufflevector(lo, hi, 0,1,2,3,4,5,6,7,8,9,10,11,12,13,14,15);
}

__device__ __forceinline__ v8f wmma16(v16h a, v16h b, v8f c) {
    return __builtin_amdgcn_wmma_f32_16x16x32_f16(false, a, false, b, (short)0, c,
                                                  false, false);
}

// ---------------------------------------------------------------------------
// K0: weights (C x 256 f32) -> f16 transposed Wt[n][c]
__global__ void prep_w(const float* __restrict__ Wq, const float* __restrict__ Wk,
                       const float* __restrict__ Wv, _Float16* __restrict__ wt) {
    int idx = blockIdx.x * 256 + threadIdx.x;   // 0 .. 3*131072-1
    int mm  = idx >> 17;
    int rem = idx & 131071;
    int c   = rem >> 8;
    int n   = rem & 255;
    const float* W = (mm == 0) ? Wq : (mm == 1) ? Wk : Wv;
    wt[(size_t)mm * 131072 + (size_t)n * C_ + c] = (_Float16)W[(size_t)c * 256 + n];
}

// ---------------------------------------------------------------------------
// K1: x(B,C,T) f32 -> xh(B,T,C) f16 (LDS-tiled transpose) and copy x into
// out channels [0, 512).
__global__ void prep_x(const float* __restrict__ x, float* __restrict__ out,
                       _Float16* __restrict__ xh) {
    __shared__ _Float16 tile[32][33];
    int b  = blockIdx.z;
    int c0 = blockIdx.y * 32;
    int t0 = blockIdx.x * 32;
    int tx = threadIdx.x;            // 0..31
    int ty = threadIdx.y;            // 0..7
    const float* xb = x   + (size_t)b * C_ * T_;
    float*       ob = out + (size_t)b * (C_ + V_) * T_;
#pragma unroll
    for (int j = 0; j < 4; ++j) {
        int c = c0 + ty + 8 * j;
        float v = xb[(size_t)c * T_ + t0 + tx];
        ob[(size_t)c * T_ + t0 + tx] = v;      // out[:, :512, :] == x
        tile[ty + 8 * j][tx] = (_Float16)v;
    }
    __syncthreads();
    _Float16* xhb = xh + (size_t)b * T_ * C_;
#pragma unroll
    for (int j = 0; j < 4; ++j) {
        int t = t0 + ty + 8 * j;
        xhb[(size_t)t * C_ + c0 + tx] = tile[tx][ty + 8 * j];
    }
}

// ---------------------------------------------------------------------------
// K2: q/k/v projections. One block = 16 t-rows, 8 waves; each wave owns 6 of
// the 48 (matrix, 16-col) output tiles. q is pre-scaled by 1/sqrt(K)=1/16.
// All 6 B fragments of a k-step are loaded before the 6-WMMA chain so the
// loads issue as one clause and overlap the MMAs.
__global__ void qkv_proj(const _Float16* __restrict__ xh, const _Float16* __restrict__ wt,
                         const float* __restrict__ bq, const float* __restrict__ bk,
                         const float* __restrict__ bv,
                         _Float16* __restrict__ qh, _Float16* __restrict__ kh,
                         _Float16* __restrict__ vT) {
    int g    = blockIdx.x * 16;          // global row in flattened (B*T)
    int wave = threadIdx.x >> 5;
    int lane = threadIdx.x & 31;
    int idx  = lane & 15;
    int h    = lane >> 4;
    int jbase = wave * 6;

    // Hoisted weight-row bases for this wave's 6 tiles
    const _Float16* wrow[6];
#pragma unroll
    for (int i = 0; i < 6; ++i) {
        int j   = jbase + i;
        int mmx = j >> 4;
        int n0  = (j & 15) * 16;
        wrow[i] = wt + (size_t)mmx * 131072 + (size_t)(n0 + idx) * C_;
    }
    const _Float16* arow = xh + (size_t)(g + idx) * C_;

    v8f acc[6];
#pragma unroll
    for (int i = 0; i < 6; ++i) acc[i] = (v8f){0.f,0.f,0.f,0.f,0.f,0.f,0.f,0.f};

    for (int kc = 0; kc < 16; ++kc) {
        int c0 = kc * 32;
        v16h a = ldfrag(arow + c0, h);
        v16h bf[6];
#pragma unroll
        for (int i = 0; i < 6; ++i) bf[i] = ldfrag(wrow[i] + c0, h);
#pragma unroll
        for (int i = 0; i < 6; ++i) acc[i] = wmma16(a, bf[i], acc[i]);
    }

    int b = g >> 12, t0 = g & (T_ - 1);
#pragma unroll
    for (int i = 0; i < 6; ++i) {
        int j    = jbase + i;
        int mmx  = j >> 4;
        int ncol = (j & 15) * 16 + idx;
        if (mmx == 0) {
            float bias = bq[ncol];
#pragma unroll
            for (int r = 0; r < 8; ++r)
                qh[(size_t)(g + r + 8 * h) * K_ + ncol] =
                    (_Float16)((acc[i][r] + bias) * 0.0625f);
        } else if (mmx == 1) {
            float bias = bk[ncol];
#pragma unroll
            for (int r = 0; r < 8; ++r)
                kh[(size_t)(g + r + 8 * h) * K_ + ncol] =
                    (_Float16)(acc[i][r] + bias);
        } else {
            float bias = bv[ncol];
            v8h pack;
#pragma unroll
            for (int r = 0; r < 8; ++r) pack[r] = (_Float16)(acc[i][r] + bias);
            *(v8h*)(vT + ((size_t)(b * V_ + ncol)) * T_ + t0 + 8 * h) = pack;
        }
    }
}

// ---------------------------------------------------------------------------
// K3: causal flash attention. One wave per 16-row q tile; s processed 32 at a
// time. Only ~1 wave/SIMD runs, so latency hiding must come from intra-wave
// MLP: all K fragments load in one deep clause before the logit WMMA chain,
// and all V fragments are issued BEFORE the softmax (they only depend on s0),
// so they are in flight during exp/shuffles and the LDS P-relayout.
__global__ void attn(const _Float16* __restrict__ qh, const _Float16* __restrict__ kh,
                     const _Float16* __restrict__ vT, float* __restrict__ out) {
    __shared__ __align__(16) _Float16 pst[2][16][40];
    int wave = threadIdx.x >> 5;
    int lane = threadIdx.x & 31;
    int idx  = lane & 15;
    int h    = lane >> 4;
    int wid  = blockIdx.x * 2 + wave;
    int g    = wid * 16;
    int b    = g >> 12, t0 = g & (T_ - 1);

    // Preload Q A-fragments (q already scaled by 1/16)
    v16h Aq[8];
#pragma unroll
    for (int kc = 0; kc < 8; ++kc)
        Aq[kc] = ldfrag(qh + (size_t)(g + idx) * K_ + kc * 32, h);

    v8f acc[16];
#pragma unroll
    for (int t = 0; t < 16; ++t) acc[t] = (v8f){0.f,0.f,0.f,0.f,0.f,0.f,0.f,0.f};
    float mrun[8], lrun[8];
#pragma unroll
    for (int r = 0; r < 8; ++r) { mrun[r] = -1e30f; lrun[r] = 0.f; }

    const _Float16* khb = kh + (size_t)b * T_ * K_;
    const _Float16* vtb = vT + (size_t)b * V_ * T_;

    for (int s0 = 0; s0 <= t0 + 15; s0 += 32) {
        // ---- K fragments: one deep load clause (32x b128)
        v16h Bk[16];
#pragma unroll
        for (int kc = 0; kc < 8; ++kc) {
            int c0 = kc * 32;
            Bk[2 * kc]     = ldfrag(khb + (size_t)(s0 + idx) * K_ + c0, h);
            Bk[2 * kc + 1] = ldfrag(khb + (size_t)(s0 + 16 + idx) * K_ + c0, h);
        }
        // ---- logits: S(16x32) = Q(16x256) @ K^T
        v8f S0 = (v8f){0.f,0.f,0.f,0.f,0.f,0.f,0.f,0.f};
        v8f S1 = S0;
#pragma unroll
        for (int kc = 0; kc < 8; ++kc) {
            S0 = wmma16(Aq[kc], Bk[2 * kc],     S0);
            S1 = wmma16(Aq[kc], Bk[2 * kc + 1], S1);
        }

        // ---- V fragments: issue now (independent of softmax), consumed later
        v16h Bv[16];
#pragma unroll
        for (int nv = 0; nv < 16; ++nv)
            Bv[nv] = ldfrag(vtb + (size_t)(nv * 16 + idx) * T_ + s0, h);

        // prefetch next K tile while we do softmax
        if (s0 + 32 <= t0 + 15)
            __builtin_prefetch(khb + (size_t)(s0 + 32 + idx) * K_, 0, 3);

        // ---- causal mask (only near diagonal)
        if (s0 + 31 > t0) {
#pragma unroll
            for (int r = 0; r < 8; ++r) {
                int trow = t0 + r + 8 * h;
                if (s0 + idx      > trow) S0[r] = -1e30f;
                if (s0 + 16 + idx > trow) S1[r] = -1e30f;
            }
        }

        // ---- online softmax (rows live across the 16 lanes of each half-wave)
        float alpha[8], P0[8], P1[8];
#pragma unroll
        for (int r = 0; r < 8; ++r) {
            float v = fmaxf(S0[r], S1[r]);
            v = fmaxf(v, __shfl_xor(v, 1, 16));
            v = fmaxf(v, __shfl_xor(v, 2, 16));
            v = fmaxf(v, __shfl_xor(v, 4, 16));
            v = fmaxf(v, __shfl_xor(v, 8, 16));
            float mn = fmaxf(mrun[r], v);
            alpha[r] = __expf(mrun[r] - mn);
            mrun[r]  = mn;
            P0[r] = __expf(S0[r] - mn);
            P1[r] = __expf(S1[r] - mn);
            float rs = P0[r] + P1[r];
            rs += __shfl_xor(rs, 1, 16);
            rs += __shfl_xor(rs, 2, 16);
            rs += __shfl_xor(rs, 4, 16);
            rs += __shfl_xor(rs, 8, 16);
            lrun[r] = lrun[r] * alpha[r] + rs;
        }
#pragma unroll
        for (int t = 0; t < 16; ++t)
#pragma unroll
            for (int r = 0; r < 8; ++r)
                acc[t][r] *= alpha[r];

        // ---- stage P (D-layout) into LDS, reload as A-fragment
#pragma unroll
        for (int r = 0; r < 8; ++r) {
            pst[wave][r + 8 * h][idx]      = (_Float16)P0[r];
            pst[wave][r + 8 * h][16 + idx] = (_Float16)P1[r];
        }
        v8h plo = *(const v8h*)(&pst[wave][idx][h * 8]);
        v8h phi = *(const v8h*)(&pst[wave][idx][16 + h * 8]);
        v16h Ap = __builtin_shufflevector(plo, phi,
                                          0,1,2,3,4,5,6,7,8,9,10,11,12,13,14,15);

        // ---- acc += P @ V over all 256 V columns (fragments already resident)
#pragma unroll
        for (int nv = 0; nv < 16; ++nv)
            acc[nv] = wmma16(Ap, Bv[nv], acc[nv]);
    }

    // ---- normalize and store out channels [512, 768): contiguous along t
    float inv[8];
#pragma unroll
    for (int r = 0; r < 8; ++r) inv[r] = 1.0f / lrun[r];
    float* ob = out + (size_t)b * (C_ + V_) * T_;
#pragma unroll
    for (int nv = 0; nv < 16; ++nv) {
        int ch = C_ + nv * 16 + idx;
        float4 lo4, hi4;
        lo4.x = acc[nv][0] * inv[0]; lo4.y = acc[nv][1] * inv[1];
        lo4.z = acc[nv][2] * inv[2]; lo4.w = acc[nv][3] * inv[3];
        hi4.x = acc[nv][4] * inv[4]; hi4.y = acc[nv][5] * inv[5];
        hi4.z = acc[nv][6] * inv[6]; hi4.w = acc[nv][7] * inv[7];
        float* dst = ob + (size_t)ch * T_ + t0 + 8 * h;
        *(float4*)dst       = lo4;
        *(float4*)(dst + 4) = hi4;
    }
}

// ---------------------------------------------------------------------------
extern "C" void kernel_launch(void* const* d_in, const int* in_sizes, int n_in,
                              void* d_out, int out_size, void* d_ws, size_t ws_size,
                              hipStream_t stream) {
    (void)in_sizes; (void)n_in; (void)out_size; (void)ws_size;
    const float* x  = (const float*)d_in[0];
    const float* Wq = (const float*)d_in[1];
    const float* bq = (const float*)d_in[2];
    const float* Wk = (const float*)d_in[3];
    const float* bk = (const float*)d_in[4];
    const float* Wv = (const float*)d_in[5];
    const float* bv = (const float*)d_in[6];
    float* out = (float*)d_out;

    char* ws = (char*)d_ws;
    _Float16* wt = (_Float16*)(ws + WT_OFF);
    _Float16* qh = (_Float16*)(ws + QH_OFF);
    _Float16* kh = (_Float16*)(ws + KH_OFF);
    _Float16* vT = (_Float16*)(ws + VT_OFF);
    _Float16* xh = (_Float16*)(ws + XH_OFF);

    prep_w<<<(3 * 256 * 512) / 256, 256, 0, stream>>>(Wq, Wk, Wv, wt);
    prep_x<<<dim3(T_ / 32, C_ / 32, B_), dim3(32, 8), 0, stream>>>(x, out, xh);
    qkv_proj<<<(B_ * T_) / 16, 256, 0, stream>>>(xh, wt, bq, bk, bv, qh, kh, vT);
    attn<<<(B_ * T_) / 32, 64, 0, stream>>>(qh, kh, vT, out);
}
